// standard_transformer_52364241273484
// MI455X (gfx1250) — matrix operand order repaired and checked by reading, not compiled, not run on hardware
//
#include <hip/hip_runtime.h>

// ---------------------------------------------------------------------------
// MI455X (gfx1250) dual-stream attention stack.
// One workgroup per (stream, batch) element; activations resident in LDS
// (<=282KB of 320KB/WGP); all matmuls via v_wmma_f32_16x16x32_bf16 (f32 acc).
// Weights prepacked (transpose + fp32->bf16) into d_ws so both WMMA operands
// load as contiguous 16B chunks per lane. Register-blocked 2x4 (32x64) jobs:
// 2 A-frags + 4 B-frags per K-step feed 8 WMMAs => 24B/lane/WMMA operand
// traffic (vs 64B unblocked) -- halves L2 weight re-reads, which roofline
// says dominate (~70GB of L2 traffic unblocked vs ~200us of WMMA compute).
// Assumes ws_size >= ~192MB.
// ---------------------------------------------------------------------------

typedef __bf16 bf16_t;
typedef __attribute__((ext_vector_type(16))) __bf16 v16bf;
typedef __attribute__((ext_vector_type(8)))  __bf16 v8bf;
typedef __attribute__((ext_vector_type(8)))  float  v8f;

#define WAVES 8
#define BLOCK (WAVES * 32)

__device__ __forceinline__ v8f wmma_bf16(v16bf a, v16bf b, v8f c) {
  return __builtin_amdgcn_wmma_f32_16x16x32_bf16(false, a, false, b, (short)0, c,
                                                 false, false);
}

// A fragment (16x32 bf16) from a row-major buffer, row stride ld (elements).
__device__ __forceinline__ v16bf load_frag_a(const bf16_t* __restrict__ base,
                                             int row0, int k0, int ld, int lane) {
  union { v16bf v; v8bf h[2]; } f;
  const int half = (lane >> 4) & 1;
  const bf16_t* p = base + (size_t)(row0 + (lane & 15)) * ld + k0 + half * 8;
  f.h[0] = *reinterpret_cast<const v8bf*>(p);
  f.h[1] = *reinterpret_cast<const v8bf*>(p + 16);
  return f.v;
}

// B fragment (32x16 bf16): data stored "n-major" (buf[n][k] contiguous in k).
__device__ __forceinline__ v16bf load_frag_b(const bf16_t* __restrict__ base,
                                             int k0, int n0, int ld, int lane) {
  union { v16bf v; v8bf h[2]; } f;
  const int half = (lane >> 4) & 1;
  const bf16_t* p = base + (size_t)(n0 + (lane & 15)) * ld + k0 + half * 8;
  f.h[0] = *reinterpret_cast<const v8bf*>(p);
  f.h[1] = *reinterpret_cast<const v8bf*>(p + 16);
  return f.v;
}

// Register-blocked MR x NC tile-strip accumulation over K.
// A: row-major [*, lda] starting at arow0; B: n-major [*, ldb] cols bn0..
// PF: prefetch B rows (only useful when B is in global memory).
template <int MR, int NC, bool PF>
__device__ __forceinline__ void mma_block(const bf16_t* __restrict__ Ab, int arow0,
                                          int lda, const bf16_t* __restrict__ Bb,
                                          int bn0, int ldb, int klen, int lane,
                                          v8f (&acc)[MR][NC]) {
  const int ln = lane & 15;
  for (int k0 = 0; k0 < klen; k0 += 32) {
    if (PF) {
#pragma unroll
      for (int n = 0; n < NC; ++n)
        __builtin_prefetch(Bb + (size_t)(bn0 + 16 * n + ln) * ldb + k0 + 64, 0, 1);
    }
    v16bf a[MR], b[NC];
#pragma unroll
    for (int m = 0; m < MR; ++m) a[m] = load_frag_a(Ab, arow0 + 16 * m, k0, lda, lane);
#pragma unroll
    for (int n = 0; n < NC; ++n) b[n] = load_frag_b(Bb, k0, bn0 + 16 * n, ldb, lane);
#pragma unroll
    for (int m = 0; m < MR; ++m)
#pragma unroll
      for (int n = 0; n < NC; ++n) acc[m][n] = wmma_bf16(a[m], b[n], acc[m][n]);
  }
}

// One-time weight prepack: W[l][i][o] fp32 -> WT[l][o][i] bf16 (tiled via LDS).
__global__ __launch_bounds__(256) void transpose_pack_kernel(
    const float* __restrict__ W, bf16_t* __restrict__ WT, int Dm) {
  __shared__ float tile[32][33];
  const int l = blockIdx.z;
  const int i0 = blockIdx.y * 32, o0 = blockIdx.x * 32;
  const float* Wl = W + (size_t)l * Dm * Dm;
  bf16_t* WTl = WT + (size_t)l * Dm * Dm;
  const int tx = threadIdx.x, ty = threadIdx.y;
  for (int r = ty; r < 32; r += 8)
    tile[r][tx] = Wl[(size_t)(i0 + r) * Dm + o0 + tx];
  __syncthreads();
  for (int r = ty; r < 32; r += 8)
    WTl[(size_t)(o0 + r) * Dm + i0 + tx] = (bf16_t)tile[tx][r];
}

// ---------------------------------------------------------------------------
// Full L-layer attention stack for one batch element per workgroup.
// ---------------------------------------------------------------------------
template <int N, int NP, int D, int DK>
__global__ __launch_bounds__(BLOCK, 1) void attn_stack_kernel(
    const float* __restrict__ x_in,
    const bf16_t* __restrict__ WqT, const bf16_t* __restrict__ WkT,
    const bf16_t* __restrict__ WvT, const bf16_t* __restrict__ WfT,
    const float* __restrict__ Bq, const float* __restrict__ Bk,
    const float* __restrict__ Bv, const float* __restrict__ Bf,
    bf16_t* __restrict__ Osc, float* __restrict__ out, int L) {
  constexpr int H     = D / DK;
  constexpr int NTR   = NP / 16;            // query row tiles (even: 10 or 6)
  constexpr int NTJ   = NP / 16;            // key tiles
  constexpr int NTC_K = DK / 16;            // head-dim column tiles (4 or 8)
  constexpr int NTC_D = D / 16;             // model-dim column tiles (32 or 64)
  constexpr int NRP   = NTR / 2;            // row-pair strips
  constexpr int NCQ   = (NTC_K >= 4) ? 4 : NTC_K;  // QKV col-group width
  constexpr int NGQ   = NTC_K / NCQ;        // QKV col groups (1 or 2)
  constexpr int NGF   = NTC_D / 4;          // Wf col groups (8 or 16)
  constexpr int NGV   = NTC_K / 2;          // P@V col groups (2 or 4)

  extern __shared__ char smem_raw[];
  bf16_t* Xs  = reinterpret_cast<bf16_t*>(smem_raw);  // [NP][D]
  bf16_t* Qs  = Xs + (size_t)NP * D;                  // [NP][DK]
  bf16_t* Ks  = Qs + (size_t)NP * DK;                 // [NP][DK]
  bf16_t* Vts = Ks + (size_t)NP * DK;                 // [DK][NP] (transposed)
  bf16_t* Ps  = Vts + (size_t)DK * NP;                // [NP][NP] probabilities

  const int b    = blockIdx.x;
  const int tid  = threadIdx.x;
  const int lane = tid & 31, wave = tid >> 5;
  const int half = (lane >> 4) & 1, ln = lane & 15;

  // Load x (fp32 -> bf16), zero padded rows once (pads stay finite forever).
  const float* xb = x_in + (size_t)b * N * D;
  for (int idx = tid; idx < NP * D; idx += BLOCK) {
    const int r = idx / D, c = idx % D;
    Xs[idx] = (bf16_t)(r < N ? xb[(size_t)r * D + c] : 0.0f);
  }
  __syncthreads();

  bf16_t* Ob = Osc + (size_t)b * NP * D;  // per-block O scratch (L2 resident)
  const float scale = 1.0f / sqrtf((float)DK);

  for (int l = 0; l < L; ++l) {
    const bf16_t* wqT = WqT + (size_t)l * D * D;
    const bf16_t* wkT = WkT + (size_t)l * D * D;
    const bf16_t* wvT = WvT + (size_t)l * D * D;
    const bf16_t* wfT = WfT + (size_t)l * D * D;
    const float* bq = Bq + (size_t)l * D;
    const float* bk = Bk + (size_t)l * D;
    const float* bv = Bv + (size_t)l * D;
    const float* bo = Bf + (size_t)l * D;

    for (int h = 0; h < H; ++h) {
      const int c0h = h * DK;

      // ---- Q/K/V head projections: 2x NCQ blocked, [NP,D] x [D,DK] + bias --
      for (int job = wave; job < 3 * NRP * NGQ; job += WAVES) {
        const int which = job / (NRP * NGQ);
        const int t = job % (NRP * NGQ);
        const int rp = t / NGQ, cg = t % NGQ;
        const int row0 = rp * 32, colL = cg * 16 * NCQ, colG = c0h + colL;
        const bf16_t* wT = (which == 0) ? wqT : (which == 1) ? wkT : wvT;
        const float* bias = (which == 0) ? bq : (which == 1) ? bk : bv;
        v8f acc[2][NCQ];
#pragma unroll
        for (int n = 0; n < NCQ; ++n) {
          const float bval = bias[colG + 16 * n + ln];
#pragma unroll
          for (int m = 0; m < 2; ++m)
#pragma unroll
            for (int v = 0; v < 8; ++v) acc[m][n][v] = bval;
        }
        mma_block<2, NCQ, true>(Xs, row0, D, wT, colG, D, D, lane, acc);
        if (which == 2) {  // V stored transposed -> Vt[f][j]
#pragma unroll
          for (int m = 0; m < 2; ++m)
#pragma unroll
            for (int n = 0; n < NCQ; ++n)
#pragma unroll
              for (int v = 0; v < 8; ++v)
                Vts[(size_t)(colL + 16 * n + ln) * NP + row0 + 16 * m + v + half * 8] =
                    (bf16_t)acc[m][n][v];
        } else {
          bf16_t* dst = (which == 0) ? Qs : Ks;
#pragma unroll
          for (int m = 0; m < 2; ++m)
#pragma unroll
            for (int n = 0; n < NCQ; ++n)
#pragma unroll
              for (int v = 0; v < 8; ++v)
                dst[(size_t)(row0 + 16 * m + v + half * 8) * DK + colL + 16 * n + ln] =
                    (bf16_t)acc[m][n][v];
        }
      }
      __syncthreads();

      // ---- S = Q K^T (scaled), masked softmax over keys, -> Ps (bf16) ----
      for (int s = wave; s < NTR; s += WAVES) {
        const int row0 = s * 16;
        v8f sc[NTJ];
#pragma unroll
        for (int jt = 0; jt < NTJ; ++jt)
#pragma unroll
          for (int v = 0; v < 8; ++v) sc[jt][v] = 0.0f;
        for (int k0 = 0; k0 < DK; k0 += 32) {
          v16bf a = load_frag_a(Qs, row0, k0, DK, lane);
#pragma unroll
          for (int jt = 0; jt < NTJ; ++jt) {
            // K row-major == K^T n-major: contiguous B fragment, no transpose.
            v16bf bb = load_frag_b(Ks, k0, jt * 16, DK, lane);
            sc[jt] = wmma_bf16(a, bb, sc[jt]);
          }
        }
#pragma unroll
        for (int jt = 0; jt < NTJ; ++jt) {  // scale + mask padded keys
          const bool valid = (jt * 16 + ln) < N;
#pragma unroll
          for (int v = 0; v < 8; ++v)
            sc[jt][v] = valid ? sc[jt][v] * scale : -1.0e30f;
        }
        float mx[8], sm[8];
#pragma unroll
        for (int v = 0; v < 8; ++v) mx[v] = -1.0e30f;
#pragma unroll
        for (int jt = 0; jt < NTJ; ++jt)
#pragma unroll
          for (int v = 0; v < 8; ++v) mx[v] = fmaxf(mx[v], sc[jt][v]);
#pragma unroll
        for (int off = 8; off >= 1; off >>= 1)
#pragma unroll
          for (int v = 0; v < 8; ++v)
            mx[v] = fmaxf(mx[v], __shfl_xor(mx[v], off));
#pragma unroll
        for (int v = 0; v < 8; ++v) sm[v] = 0.0f;
#pragma unroll
        for (int jt = 0; jt < NTJ; ++jt)
#pragma unroll
          for (int v = 0; v < 8; ++v) {
            const float e = __expf(sc[jt][v] - mx[v]);
            sc[jt][v] = e;
            sm[v] += e;
          }
#pragma unroll
        for (int off = 8; off >= 1; off >>= 1)
#pragma unroll
          for (int v = 0; v < 8; ++v) sm[v] += __shfl_xor(sm[v], off);
        float inv[8];
#pragma unroll
        for (int v = 0; v < 8; ++v) inv[v] = 1.0f / sm[v];
#pragma unroll
        for (int jt = 0; jt < NTJ; ++jt)  // store probs row-major (A layout)
#pragma unroll
          for (int v = 0; v < 8; ++v)
            Ps[(size_t)(row0 + v + half * 8) * NP + jt * 16 + ln] =
                (bf16_t)(sc[jt][v] * inv[v]);
      }
      __syncthreads();

      // ---- O_h = P @ V : 2x2 blocked, LDS x LDS -> global O slice ----
      for (int job = wave; job < NRP * NGV; job += WAVES) {
        const int rp = job / NGV, cg = job % NGV;
        const int row0 = rp * 32, f0 = cg * 32;
        v8f acc[2][2];
#pragma unroll
        for (int m = 0; m < 2; ++m)
#pragma unroll
          for (int n = 0; n < 2; ++n)
#pragma unroll
            for (int v = 0; v < 8; ++v) acc[m][n][v] = 0.0f;
        mma_block<2, 2, false>(Ps, row0, NP, Vts, f0, NP, NP, lane, acc);
#pragma unroll
        for (int m = 0; m < 2; ++m)
#pragma unroll
          for (int n = 0; n < 2; ++n)
#pragma unroll
            for (int v = 0; v < 8; ++v)
              Ob[(size_t)(row0 + 16 * m + v + half * 8) * D + c0h + f0 + 16 * n + ln] =
                  (bf16_t)acc[m][n][v];
      }
      __syncthreads();
    }  // heads

    __threadfence();  // O stores visible to all waves' global loads
    __syncthreads();

    // ---- x = O @ Wf + bf : 2x4 blocked, [NP,D] x [D,D], overwrite Xs ----
    for (int job = wave; job < NRP * NGF; job += WAVES) {
      const int rp = job / NGF, cg = job % NGF;
      const int row0 = rp * 32, col0 = cg * 64;
      v8f acc[2][4];
#pragma unroll
      for (int n = 0; n < 4; ++n) {
        const float bval = bo[col0 + 16 * n + ln];
#pragma unroll
        for (int m = 0; m < 2; ++m)
#pragma unroll
          for (int v = 0; v < 8; ++v) acc[m][n][v] = bval;
      }
      mma_block<2, 4, true>(Ob, row0, D, wfT, col0, D, D, lane, acc);
#pragma unroll
      for (int m = 0; m < 2; ++m)
#pragma unroll
        for (int n = 0; n < 4; ++n)
#pragma unroll
          for (int v = 0; v < 8; ++v)
            Xs[(size_t)(row0 + 16 * m + v + half * 8) * D + col0 + 16 * n + ln] =
                (bf16_t)acc[m][n][v];
    }
    __syncthreads();
  }  // layers

  // Emit class token (row 0) as fp32.
  for (int c = tid; c < D; c += BLOCK)
    out[(size_t)b * D + c] = (float)Xs[c];
}

// ---------------------------------------------------------------------------
extern "C" void kernel_launch(void* const* d_in, const int* in_sizes, int n_in,
                              void* d_out, int out_size, void* d_ws,
                              size_t ws_size, hipStream_t stream) {
  (void)in_sizes; (void)n_in; (void)out_size; (void)ws_size;
  const float* xl   = (const float*)d_in[0];
  const float* xs   = (const float*)d_in[1];
  const float* Wq_l = (const float*)d_in[2];  const float* bq_l = (const float*)d_in[3];
  const float* Wk_l = (const float*)d_in[4];  const float* bk_l = (const float*)d_in[5];
  const float* Wv_l = (const float*)d_in[6];  const float* bv_l = (const float*)d_in[7];
  const float* Wf_l = (const float*)d_in[8];  const float* bf_l = (const float*)d_in[9];
  const float* Wq_s = (const float*)d_in[10]; const float* bq_s = (const float*)d_in[11];
  const float* Wk_s = (const float*)d_in[12]; const float* bk_s = (const float*)d_in[13];
  const float* Wv_s = (const float*)d_in[14]; const float* bv_s = (const float*)d_in[15];
  const float* Wf_s = (const float*)d_in[16]; const float* bf_s = (const float*)d_in[17];

  // Workspace carve-out (~191 MB): bf16 transposed weights + O scratch.
  char* ws = (char*)d_ws;
  size_t off = 0;
  auto carve = [&](size_t bytes) -> char* {
    char* p = ws + off;
    off += (bytes + 255) & ~(size_t)255;
    return p;
  };
  const size_t szL = (size_t)16 * 512 * 512 * sizeof(bf16_t);
  const size_t szS = (size_t)16 * 1024 * 1024 * sizeof(bf16_t);
  bf16_t* WqTl = (bf16_t*)carve(szL); bf16_t* WkTl = (bf16_t*)carve(szL);
  bf16_t* WvTl = (bf16_t*)carve(szL); bf16_t* WfTl = (bf16_t*)carve(szL);
  bf16_t* WqTs = (bf16_t*)carve(szS); bf16_t* WkTs = (bf16_t*)carve(szS);
  bf16_t* WvTs = (bf16_t*)carve(szS); bf16_t* WfTs = (bf16_t*)carve(szS);
  bf16_t* Ol = (bf16_t*)carve((size_t)64 * 160 * 512 * sizeof(bf16_t));
  bf16_t* Os = (bf16_t*)carve((size_t)64 * 96 * 1024 * sizeof(bf16_t));

  // One-time (per call) weight transpose + fp32->bf16 pack.
  dim3 tb(32, 8, 1);
  transpose_pack_kernel<<<dim3(16, 16, 16), tb, 0, stream>>>(Wq_l, WqTl, 512);
  transpose_pack_kernel<<<dim3(16, 16, 16), tb, 0, stream>>>(Wk_l, WkTl, 512);
  transpose_pack_kernel<<<dim3(16, 16, 16), tb, 0, stream>>>(Wv_l, WvTl, 512);
  transpose_pack_kernel<<<dim3(16, 16, 16), tb, 0, stream>>>(Wf_l, WfTl, 512);
  transpose_pack_kernel<<<dim3(32, 32, 16), tb, 0, stream>>>(Wq_s, WqTs, 1024);
  transpose_pack_kernel<<<dim3(32, 32, 16), tb, 0, stream>>>(Wk_s, WkTs, 1024);
  transpose_pack_kernel<<<dim3(32, 32, 16), tb, 0, stream>>>(Wv_s, WvTs, 1024);
  transpose_pack_kernel<<<dim3(32, 32, 16), tb, 0, stream>>>(Wf_s, WfTs, 1024);

  // Dynamic LDS: large 276,480 B; small 288,768 B (<= 320KB/WGP).
  const int SMEM_L = (160 * 512 + 3 * 160 * 64 + 160 * 160) * (int)sizeof(bf16_t);
  const int SMEM_S = (96 * 1024 + 3 * 96 * 128 + 96 * 96) * (int)sizeof(bf16_t);
  hipFuncSetAttribute(reinterpret_cast<const void*>(attn_stack_kernel<129, 160, 512, 64>),
                      hipFuncAttributeMaxDynamicSharedMemorySize, SMEM_L);
  hipFuncSetAttribute(reinterpret_cast<const void*>(attn_stack_kernel<65, 96, 1024, 128>),
                      hipFuncAttributeMaxDynamicSharedMemorySize, SMEM_S);

  float* out = (float*)d_out;
  attn_stack_kernel<129, 160, 512, 64><<<64, BLOCK, SMEM_L, stream>>>(
      xl, WqTl, WkTl, WvTl, WfTl, bq_l, bk_l, bv_l, bf_l, Ol, out, 16);
  attn_stack_kernel<65, 96, 1024, 128><<<64, BLOCK, SMEM_S, stream>>>(
      xs, WqTs, WkTs, WvTs, WfTs, bq_s, bk_s, bv_s, bf_s, Os, out + 64 * 512, 16);
}
